// Retention_41970420416516
// MI455X (gfx1250) — matrix operand order repaired and checked by reading, not compiled, not run on hardware
//
#include <hip/hip_runtime.h>
#include <hip/hip_bf16.h>

// Retention: out[b,i,:] = (1/sqrt(d)) * q_i * S_i,  S_i = a*S_{i-1} + k_i v_i^T
// Chunked linear recurrence with bf16 WMMA + async global->LDS staging (gfx1250).

typedef __bf16 bf16_t;
typedef __attribute__((ext_vector_type(16))) __bf16 v16bf;
typedef __attribute__((ext_vector_type(8)))  __bf16 v8bf;
typedef __attribute__((ext_vector_type(8)))  float  v8f;

#define SEQ    4096
#define DIM    256
#define BATCH  8
#define CHUNK  64
#define NCHUNK (SEQ / CHUNK)
#define L2A    (-0.014499569695115089f)   // log2(0.99)
#define A64F   (0.5255964875f)            // 0.99^64

__device__ __forceinline__ v8f wmma_bf16(v16bf a, v16bf b, v8f c) {
  return __builtin_amdgcn_wmma_f32_16x16x32_bf16(false, a, false, b, (short)0, c,
                                                 false, false);
}

// A-fragment (16x32 bf16): lane = row (lane&15); half=lane>>4 selects K offset.
__device__ __forceinline__ v16bf load_a_frag(const bf16_t* base, int row_stride,
                                             int kbase) {
  const int lane = threadIdx.x & 31;
  const int hf = lane >> 4;
  const int row = lane & 15;
  const bf16_t* p = base + (size_t)row * row_stride + kbase + hf * 8;
  v8bf lo = *(const v8bf*)p;
  v8bf hi = *(const v8bf*)(p + 16);
  v16bf r;
#pragma unroll
  for (int i = 0; i < 8; ++i) { r[i] = lo[i]; r[i + 8] = hi[i]; }
  return r;
}

// B-fragment (32x16 bf16): lane = col; source layout [col][K] contiguous in K.
__device__ __forceinline__ v16bf load_b_frag(const bf16_t* base, int col_stride,
                                             int kbase) {
  const int lane = threadIdx.x & 31;
  const int hf = lane >> 4;
  const int col = lane & 15;
  const bf16_t* p = base + (size_t)col * col_stride + kbase + hf * 8;
  v8bf lo = *(const v8bf*)p;
  v8bf hi = *(const v8bf*)(p + 16);
  v16bf r;
#pragma unroll
  for (int i = 0; i < 8; ++i) { r[i] = lo[i]; r[i + 8] = hi[i]; }
  return r;
}

// Async 16B global -> LDS copy (CDNA5 ASYNCcnt path).
__device__ __forceinline__ void async_copy_b128(bf16_t* lds_dst,
                                                const bf16_t* gsrc) {
  const unsigned lds_addr = (unsigned)(uintptr_t)lds_dst;  // low 32 = LDS offset
  const unsigned long long ga = (unsigned long long)(uintptr_t)gsrc;
  asm volatile("global_load_async_to_lds_b128 %0, %1, off"
               :
               : "v"(lds_addr), "v"(ga)
               : "memory");
}
__device__ __forceinline__ void wait_async0() {
  asm volatile("s_wait_asynccnt 0x0" ::: "memory");
}

// ---------------------------------------------------------------------------
// Kernel 1: qkv = x @ W + b.  Emits:
//   q   [32768][256] bf16, prescaled by 1/sqrt(256)
//   k   [32768][256] bf16                     (score B-operands)
//   vT  [8][256][4096] bf16 (v transposed)    (intra/update B-operands)
//   kts [8][256][4096] bf16 = a^(63-t%64)*k^T (update A-operands)
// Block tile 64x64, BK=32. 256 threads = 8 waves.
// ---------------------------------------------------------------------------
__global__ __launch_bounds__(256) void qkv_gemm_kernel(
    const float* __restrict__ x, const float* __restrict__ W,
    const float* __restrict__ bias, bf16_t* __restrict__ qo,
    bf16_t* __restrict__ ko, bf16_t* __restrict__ vt,
    bf16_t* __restrict__ kts) {
  __shared__ bf16_t As[64][40];  // [m][k], 80B rows (16B aligned, bank-spread)
  __shared__ bf16_t Bs[64][40];  // [n][k]

  const int row0 = blockIdx.y * 64;
  const int col0 = blockIdx.x * 64;
  const int tid = threadIdx.x;
  const int wave = tid >> 5, lane = tid & 31;
  const int hf = lane >> 4, l15 = lane & 15;
  const int wm = (wave & 3) * 16;
  const int wn = (wave >> 2) * 32;

  v8f acc0 = {}, acc1 = {};

  for (int kb = 0; kb < DIM; kb += 32) {
    {  // stage x tile 64x32 (f32 -> bf16)
      const int r = tid >> 2, c = (tid & 3) * 8;
      const float* src = x + (size_t)(row0 + r) * DIM + kb + c;
      float4 f0 = *(const float4*)src;
      float4 f1 = *(const float4*)(src + 4);
      bf16_t* dst = &As[r][c];
      dst[0] = (bf16_t)f0.x; dst[1] = (bf16_t)f0.y;
      dst[2] = (bf16_t)f0.z; dst[3] = (bf16_t)f0.w;
      dst[4] = (bf16_t)f1.x; dst[5] = (bf16_t)f1.y;
      dst[6] = (bf16_t)f1.z; dst[7] = (bf16_t)f1.w;
    }
    {  // stage W tile 32x64 transposed -> Bs[n][k]
      const int kr = tid >> 3, nc = (tid & 7) * 8;
      const float* src = W + (size_t)(kb + kr) * 768 + col0 + nc;
      float4 f0 = *(const float4*)src;
      float4 f1 = *(const float4*)(src + 4);
      Bs[nc + 0][kr] = (bf16_t)f0.x; Bs[nc + 1][kr] = (bf16_t)f0.y;
      Bs[nc + 2][kr] = (bf16_t)f0.z; Bs[nc + 3][kr] = (bf16_t)f0.w;
      Bs[nc + 4][kr] = (bf16_t)f1.x; Bs[nc + 5][kr] = (bf16_t)f1.y;
      Bs[nc + 6][kr] = (bf16_t)f1.z; Bs[nc + 7][kr] = (bf16_t)f1.w;
    }
    __syncthreads();
    v16bf a = load_a_frag(&As[wm][0], 40, 0);
    v16bf b0 = load_b_frag(&Bs[wn][0], 40, 0);
    v16bf b1 = load_b_frag(&Bs[wn + 16][0], 40, 0);
    acc0 = wmma_bf16(a, b0, acc0);
    acc1 = wmma_bf16(a, b1, acc1);
    __syncthreads();
  }

#pragma unroll
  for (int tile = 0; tile < 2; ++tile) {
    const int gcol = col0 + wn + tile * 16 + l15;
    const float bv = bias[gcol];
    const int region = gcol >> 8;  // 0=q, 1=k, 2=v
    const int lcol = gcol & 255;
    v8f a = tile ? acc1 : acc0;
    const int grow0 = row0 + wm + hf * 8;       // 8 consecutive rows => 8 consec t
    const int batch = grow0 >> 12;
    const int t0 = grow0 & (SEQ - 1);
    if (region == 0) {
#pragma unroll
      for (int r = 0; r < 8; ++r)
        qo[(size_t)(grow0 + r) * DIM + lcol] = (bf16_t)((a[r] + bv) * 0.0625f);
    } else if (region == 1) {
      v8bf pk;
#pragma unroll
      for (int r = 0; r < 8; ++r) {
        const float val = a[r] + bv;
        ko[(size_t)(grow0 + r) * DIM + lcol] = (bf16_t)val;
        const float eta =
            __builtin_exp2f((float)(CHUNK - 1 - ((t0 & 63) + r)) * L2A);
        pk[r] = (bf16_t)(val * eta);
      }
      *(v8bf*)(kts + ((size_t)batch * DIM + lcol) * SEQ + t0) = pk;
    } else {
      v8bf pk;
#pragma unroll
      for (int r = 0; r < 8; ++r) pk[r] = (bf16_t)(a[r] + bv);
      *(v8bf*)(vt + ((size_t)batch * DIM + lcol) * SEQ + t0) = pk;
    }
  }
}

// ---------------------------------------------------------------------------
// Kernel 2: chunked retention scan.
// grid = (4 col-slices, 8 batches); block = 256 threads = 8 waves.
// State R [256 x 64] f32 lives in accumulator VGPRs (8 v8f tiles per wave).
// Per chunk: async-stage Vs/KT (pure copies of pre-transposed operands),
// overlap with the masked score GEMM, then cross/intra/update WMMA GEMMs.
// ---------------------------------------------------------------------------
#define RB_S 264  // Rb row stride (bf16): [col][d]
#define T_S  72   // KT/Vs/Ss row stride (bf16)

__global__ __launch_bounds__(256) void retention_scan_kernel(
    const bf16_t* __restrict__ qg, const bf16_t* __restrict__ kg,
    const bf16_t* __restrict__ vtg, const bf16_t* __restrict__ ktsg,
    float* __restrict__ out) {
  extern __shared__ char smem_raw[];
  bf16_t* Rb = (bf16_t*)smem_raw;  // [64][RB_S]  state bf16, [col][d]
  bf16_t* KT = Rb + 64 * RB_S;     // [256][T_S]  eta*k^T, [d][t]
  bf16_t* Vs = KT + 256 * T_S;     // [64][T_S]   v^T, [col][t]
  bf16_t* Ss = Vs + 64 * T_S;      // [64][T_S]   masked scores [t][t']

  const int batch = blockIdx.y, slice = blockIdx.x;
  const int tid = threadIdx.x, wave = tid >> 5, lane = tid & 31;
  const int hf = lane >> 4, l15 = lane & 15;

  const size_t bbase = (size_t)batch * SEQ * DIM;
  const bf16_t* qb = qg + bbase;
  const bf16_t* kb = kg + bbase;
  const bf16_t* vtb = vtg + bbase + (size_t)slice * 64 * SEQ;  // [64 cols][4096]
  const bf16_t* ktb = ktsg + bbase;                            // [256 d][4096]
  float* ob = out + bbase + slice * 64;

  const int mr = wave >> 1;        // out/score row tile
  const int nc0 = (wave & 1) * 2;  // out/score col tile base
  const int rt0 = wave * 2;        // state row tiles

  v8f R[8];
#pragma unroll
  for (int i = 0; i < 8; ++i) R[i] = (v8f){};

  for (int ch = 0; ch < NCHUNK; ++ch) {
    const int cs = ch * CHUNK;
    __syncthreads();  // previous chunk's LDS consumers are done

    // ---- phase 1a: async-stage Vs (v^T slice) and KT (eta*k^T) ----
    {  // Vs[col][t]: 64 rows x 128B; 4 threads per row, 2x16B each
      const int col = tid >> 2, c = (tid & 3) * 8;
      const bf16_t* src = vtb + (size_t)col * SEQ + cs + c;
      async_copy_b128(Vs + col * T_S + c, src);
      async_copy_b128(Vs + col * T_S + c + 32, src + 32);
    }
    {  // KT[d][t]: 256 rows x 128B; 1 thread per row, 8x16B
      const bf16_t* src = ktb + (size_t)tid * SEQ + cs;
      bf16_t* dst = KT + tid * T_S;
#pragma unroll
      for (int c = 0; c < 8; ++c) async_copy_b128(dst + c * 8, src + c * 8);
    }
    // prefetch next chunk's q/k rows while we compute
    if (ch + 1 < NCHUNK) {
      __builtin_prefetch(qb + (size_t)(cs + CHUNK + (tid >> 2)) * DIM, 0, 1);
      __builtin_prefetch(kb + (size_t)(cs + CHUNK + (tid >> 2)) * DIM, 0, 1);
    }

    // ---- phase 1b: stage Rb (state -> bf16) ----
#pragma unroll
    for (int rt2 = 0; rt2 < 2; ++rt2)
#pragma unroll
      for (int ct = 0; ct < 4; ++ct) {
        v8f a = R[rt2 * 4 + ct];
        bf16_t* dst =
            Rb + (size_t)(ct * 16 + l15) * RB_S + (rt0 + rt2) * 16 + hf * 8;
#pragma unroll
        for (int r = 0; r < 8; ++r) dst[r] = (bf16_t)a[r];
      }

    // ---- phase 2: masked intra-chunk scores S = (q k^T) o D (global reads) ----
    v8f sA0 = {}, sA1 = {};
    {
      const bf16_t* qrow = qb + (size_t)(cs + mr * 16) * DIM;
      const bf16_t* k0 = kb + (size_t)(cs + nc0 * 16) * DIM;
      const bf16_t* k1 = kb + (size_t)(cs + (nc0 + 1) * 16) * DIM;
#pragma unroll
      for (int kk = 0; kk < DIM; kk += 32) {
        v16bf a = load_a_frag(qrow, DIM, kk);
        v16bf b0 = load_b_frag(k0, DIM, kk);
        v16bf b1 = load_b_frag(k1, DIM, kk);
        sA0 = wmma_bf16(a, b0, sA0);
        sA1 = wmma_bf16(a, b1, sA1);
      }
    }
#pragma unroll
    for (int r = 0; r < 8; ++r) {  // causal decay mask, bf16 store to LDS
      const int tt = mr * 16 + hf * 8 + r;
      const int tp0 = nc0 * 16 + l15, tp1 = tp0 + 16;
      const float m0 =
          (tt >= tp0) ? __builtin_exp2f((float)(tt - tp0) * L2A) : 0.0f;
      const float m1 =
          (tt >= tp1) ? __builtin_exp2f((float)(tt - tp1) * L2A) : 0.0f;
      Ss[tt * T_S + tp0] = (bf16_t)(sA0[r] * m0);
      Ss[tt * T_S + tp1] = (bf16_t)(sA1[r] * m1);
    }
    wait_async0();    // Vs/KT transfers landed
    __syncthreads();  // ... and visible to all waves (Ss/Rb too)

    // ---- phase 3a: cross term q @ Rb ----
    v8f cr0 = {}, cr1 = {};
    {
      const bf16_t* qrow = qb + (size_t)(cs + mr * 16) * DIM;
#pragma unroll
      for (int kk = 0; kk < DIM; kk += 32) {
        v16bf a = load_a_frag(qrow, DIM, kk);
        v16bf b0 = load_b_frag(Rb + (size_t)(nc0 * 16) * RB_S, RB_S, kk);
        v16bf b1 = load_b_frag(Rb + (size_t)((nc0 + 1) * 16) * RB_S, RB_S, kk);
        cr0 = wmma_bf16(a, b0, cr0);
        cr1 = wmma_bf16(a, b1, cr1);
      }
    }
    // ---- phase 3b: intra term S @ V ----
    v8f in0 = {}, in1 = {};
#pragma unroll
    for (int ks = 0; ks < 2; ++ks) {
      v16bf a = load_a_frag(Ss + (size_t)(mr * 16) * T_S, T_S, ks * 32);
      v16bf b0 = load_b_frag(Vs + (size_t)(nc0 * 16) * T_S, T_S, ks * 32);
      v16bf b1 = load_b_frag(Vs + (size_t)((nc0 + 1) * 16) * T_S, T_S, ks * 32);
      in0 = wmma_bf16(a, b0, in0);
      in1 = wmma_bf16(a, b1, in1);
    }
    // ---- phase 3c: state update R = a^64 * R + (eta k)^T @ V ----
#pragma unroll
    for (int i = 0; i < 8; ++i)
#pragma unroll
      for (int r = 0; r < 8; ++r) R[i][r] *= A64F;
#pragma unroll
    for (int ks = 0; ks < 2; ++ks) {
      v16bf bfr[4];
#pragma unroll
      for (int ct = 0; ct < 4; ++ct)
        bfr[ct] = load_b_frag(Vs + (size_t)(ct * 16) * T_S, T_S, ks * 32);
#pragma unroll
      for (int rt2 = 0; rt2 < 2; ++rt2) {
        v16bf a =
            load_a_frag(KT + (size_t)((rt0 + rt2) * 16) * T_S, T_S, ks * 32);
#pragma unroll
        for (int ct = 0; ct < 4; ++ct)
          R[rt2 * 4 + ct] = wmma_bf16(a, bfr[ct], R[rt2 * 4 + ct]);
      }
    }
    // ---- output: out = intra + alpha^(t+1) * cross ----
#pragma unroll
    for (int r = 0; r < 8; ++r) {
      const int t = mr * 16 + hf * 8 + r;
      const float xi = __builtin_exp2f((float)(t + 1) * L2A);
      const size_t row = (size_t)(cs + t) * DIM;
      ob[row + nc0 * 16 + l15] = in0[r] + xi * cr0[r];
      ob[row + (nc0 + 1) * 16 + l15] = in1[r] + xi * cr1[r];
    }
  }
}

extern "C" void kernel_launch(void* const* d_in, const int* in_sizes, int n_in,
                              void* d_out, int out_size, void* d_ws,
                              size_t ws_size, hipStream_t stream) {
  const float* x = (const float*)d_in[0];  // [8,4096,256]
  const float* W = (const float*)d_in[1];  // [256,768]
  const float* b = (const float*)d_in[2];  // [768]
  float* out = (float*)d_out;              // [8,4096,256] f32

  const size_t ROWS = (size_t)BATCH * SEQ;  // 32768
  bf16_t* q = (bf16_t*)d_ws;                // 4 x 16 MB in workspace
  bf16_t* k = q + ROWS * DIM;
  bf16_t* vt = k + ROWS * DIM;
  bf16_t* kts = vt + ROWS * DIM;

  qkv_gemm_kernel<<<dim3(768 / 64, ROWS / 64), 256, 0, stream>>>(x, W, b, q, k,
                                                                 vt, kts);

  const size_t lds_bytes =
      (size_t)(64 * RB_S + 256 * T_S + 64 * T_S + 64 * T_S) * sizeof(bf16_t);
  retention_scan_kernel<<<dim3(4, BATCH), 256, lds_bytes, stream>>>(q, k, vt,
                                                                    kts, out);
}
